// LSTMJump_20547123544848
// MI455X (gfx1250) — matrix operand order repaired and checked by reading, not compiled, not run on hardware
//
#include <hip/hip_runtime.h>
#include <stdint.h>

typedef __bf16 bf16;
typedef __attribute__((ext_vector_type(16))) __bf16 v16bf;
typedef __attribute__((ext_vector_type(8)))  __bf16 v8bf;
typedef __attribute__((ext_vector_type(8)))  float  v8f;
typedef __attribute__((ext_vector_type(4)))  int    v4i;

#define S_LEN  512
#define E_DIM  256
#define H_DIM  512
#define K_DIM  768    // E + H
#define N4H    2048   // 4*H
#define B_SZ   256
#define KP1    41     // K+1 jump options
#define NSTEPS 5

#if __has_builtin(__builtin_amdgcn_global_load_async_to_lds_b128) && \
    __has_builtin(__builtin_amdgcn_s_wait_asynccnt)
#define HAVE_ASYNC_LDS 1
#endif

__device__ __forceinline__ unsigned hashu(unsigned x) {
  x ^= x >> 16; x *= 0x7feb352dU;
  x ^= x >> 15; x *= 0x846ca68bU;
  x ^= x >> 16; return x;
}
__device__ __forceinline__ float sigf(float x) { return 1.f / (1.f + __expf(-x)); }

// ------------------------------------------------------------------ zero state
__global__ void k_zero(float* p, long n) {
  long i = (long)blockIdx.x * blockDim.x + threadIdx.x;
  if (i < n) p[i] = 0.f;
}

// --------------------------------------------- pack weights to bf16, bias sums
// Wp[n,k] (2048 x 768 bf16):  k<256 -> Wih[n,k] ; else Whh[n,k-256]
__global__ void k_init_w(const float* __restrict__ Wih, const float* __restrict__ Whh,
                         const float* __restrict__ bih, const float* __restrict__ bhh,
                         bf16* __restrict__ Wp, float* __restrict__ bsum) {
  int i = blockIdx.x * blockDim.x + threadIdx.x;
  if (i < N4H * K_DIM) {
    int n = i / K_DIM, k = i - n * K_DIM;
    float w = (k < E_DIM) ? Wih[n * E_DIM + k] : Whh[n * H_DIM + (k - E_DIM)];
    Wp[i] = (bf16)w;
  }
  if (i < N4H) bsum[i] = bih[i] + bhh[i];
}

// -------------------------------------------------------------- per-step prep:
// row clamp / hidden capture, embedding gather (+hash dropout), h -> bf16.
// embh (A matrix) layout: [B, 768] bf16 = [emb(256) | h(512)] row-major.
__global__ __launch_bounds__(256) void k_prep(
    const int* __restrict__ xs, const float* __restrict__ embed_w,
    const float* __restrict__ h, float* __restrict__ hiddens,
    const int* __restrict__ rows, int* __restrict__ rows_c,
    int* __restrict__ fpflag, int* __restrict__ captured,
    bf16* __restrict__ embh, const int* __restrict__ flags, int flagIdx) {
  const int b = blockIdx.x, tid = threadIdx.x;
  const int act = (flags[flagIdx] == 0);
  const int r = rows[b];
  const int fp = (r >= S_LEN);
  const int rc = fp ? (S_LEN - 1) : r;
  if (tid == 0) {
    rows_c[b] = rc;
    fpflag[b] = fp;
    if (fp && act) captured[b] = 1;
  }
  if (fp && act) {
    for (int j = tid; j < H_DIM; j += 256) hiddens[b * H_DIM + j] = h[b * H_DIM + j];
  }
  // embedding gather + sum over 8 codes + deterministic hash dropout(p=0.1)
  {
    const int e = tid;  // E_DIM == blockDim
    float s = 0.f;
    const int* xrow = xs + ((size_t)b * S_LEN + rc) * 8;
#pragma unroll
    for (int m = 0; m < 8; ++m) s += embed_w[(size_t)xrow[m] * E_DIM + e];
    unsigned hsh = hashu((unsigned)((b * S_LEN + rc) * E_DIM + e) ^ 0x9E3779B9u);
    s = (((hsh >> 8) & 0xFFFFFF) < 15099494u /* 0.9*2^24 */) ? s * (1.f / 0.9f) : 0.f;
    embh[(size_t)b * K_DIM + e] = (bf16)s;
  }
  for (int j = tid; j < H_DIM; j += 256)
    embh[(size_t)b * K_DIM + E_DIM + j] = (bf16)h[b * H_DIM + j];
}

// -------------------------------------------------- LSTM step GEMM + gate fuse
// grid 64 = 16 M-tiles x 4 column groups; 256 thr = 8 waves; one 16x16 h tile
// per wave, with 4 gate accumulators (i,f,g,o) -> c2/h2 computed in registers.
__global__ __launch_bounds__(256) void k_lstm_gemm(
    const bf16* __restrict__ Wp, const float* __restrict__ bsum,
    const bf16* __restrict__ embh,
    float* __restrict__ h, float* __restrict__ c,
    const int* __restrict__ flags, int flagIdx) {
  if (flags[flagIdx]) return;  // act==false: h,c unchanged this step
  __shared__ __align__(16) bf16 ldsA[16 * K_DIM];  // 24 KB A strip

  const int tid  = threadIdx.x;
  const int wave = tid >> 5;
  const int lane = tid & 31;
  const int mt   = blockIdx.x & 15;  // rows mt*16 .. mt*16+15
  const int jg   = blockIdx.x >> 4;  // 4 groups of 128 h-columns
  const int j0   = jg * 128 + wave * 16;

  // stage 16x768 bf16 A strip into LDS (contiguous 24576B block, 1536 x 16B)
  {
    const char* gsrc = (const char*)(embh + (size_t)mt * 16 * K_DIM);
#ifdef HAVE_ASYNC_LDS
    typedef __attribute__((address_space(1))) v4i* gp1_t;
    typedef __attribute__((address_space(3))) v4i* lp3_t;
    gp1_t g1 = (gp1_t)gsrc;                 // 16B units, const cast away
    lp3_t l3 = (lp3_t)(v4i*)(char*)ldsA;
#pragma unroll
    for (int i = 0; i < 6; ++i)
      __builtin_amdgcn_global_load_async_to_lds_b128(
          g1 + (tid + i * 256), l3 + (tid + i * 256), 0, 0);
    __builtin_amdgcn_s_wait_asynccnt(0);
#else
    const uint4* g4 = (const uint4*)gsrc;
    uint4* l4 = (uint4*)ldsA;
#pragma unroll
    for (int i = 0; i < 6; ++i) l4[tid + i * 256] = g4[tid + i * 256];
#endif
  }
  __syncthreads();

  v8f ai = {}, af = {}, ag = {}, ao = {};
  const int arow = lane & 15;
  const int kbA  = (lane >> 4) * 16;  // A frag: K kb16..kb16+15
  const int kbB  = (lane >> 4) * 16;  // B frag: K kb16..kb16+15
  const int ncol = j0 + (lane & 15);
  const bf16* Arow = ldsA + arow * K_DIM + kbA;
  const bf16* B0 = Wp + ((size_t)(0 * H_DIM) + ncol) * K_DIM + kbB;
  const bf16* B1 = Wp + ((size_t)(1 * H_DIM) + ncol) * K_DIM + kbB;
  const bf16* B2 = Wp + ((size_t)(2 * H_DIM) + ncol) * K_DIM + kbB;
  const bf16* B3 = Wp + ((size_t)(3 * H_DIM) + ncol) * K_DIM + kbB;

  for (int k = 0; k < K_DIM; k += 32) {
    __builtin_prefetch(B0 + k + 128, 0, 1);
    v8bf alo = *(const v8bf*)(Arow + k);
    v8bf ahi = *(const v8bf*)(Arow + k + 8);
    v16bf a = __builtin_shufflevector(alo, ahi, 0, 1, 2, 3, 4, 5, 6, 7, 8, 9, 10,
                                      11, 12, 13, 14, 15);
    {
      v8bf lo = *(const v8bf*)(B0 + k), hi = *(const v8bf*)(B0 + k + 8);
      v16bf bm = __builtin_shufflevector(lo, hi, 0, 1, 2, 3, 4, 5, 6, 7, 8, 9, 10,
                                         11, 12, 13, 14, 15);
      ai = __builtin_amdgcn_wmma_f32_16x16x32_bf16(false, a, false, bm, (short)0,
                                                   ai, false, false);
    }
    {
      v8bf lo = *(const v8bf*)(B1 + k), hi = *(const v8bf*)(B1 + k + 8);
      v16bf bm = __builtin_shufflevector(lo, hi, 0, 1, 2, 3, 4, 5, 6, 7, 8, 9, 10,
                                         11, 12, 13, 14, 15);
      af = __builtin_amdgcn_wmma_f32_16x16x32_bf16(false, a, false, bm, (short)0,
                                                   af, false, false);
    }
    {
      v8bf lo = *(const v8bf*)(B2 + k), hi = *(const v8bf*)(B2 + k + 8);
      v16bf bm = __builtin_shufflevector(lo, hi, 0, 1, 2, 3, 4, 5, 6, 7, 8, 9, 10,
                                         11, 12, 13, 14, 15);
      ag = __builtin_amdgcn_wmma_f32_16x16x32_bf16(false, a, false, bm, (short)0,
                                                   ag, false, false);
    }
    {
      v8bf lo = *(const v8bf*)(B3 + k), hi = *(const v8bf*)(B3 + k + 8);
      v16bf bm = __builtin_shufflevector(lo, hi, 0, 1, 2, 3, 4, 5, 6, 7, 8, 9, 10,
                                         11, 12, 13, 14, 15);
      ao = __builtin_amdgcn_wmma_f32_16x16x32_bf16(false, a, false, bm, (short)0,
                                                   ao, false, false);
    }
  }

  // fused gate epilogue (C/D layout: lanes 0-15 -> M=v, lanes 16-31 -> M=8+v)
  const float bi_ = bsum[0 * H_DIM + ncol];
  const float bf_ = bsum[1 * H_DIM + ncol];
  const float bg_ = bsum[2 * H_DIM + ncol];
  const float bo_ = bsum[3 * H_DIM + ncol];
  const int m0 = mt * 16 + ((lane >> 4) << 3);
#pragma unroll
  for (int v = 0; v < 8; ++v) {
    int idx = (m0 + v) * H_DIM + ncol;
    float zi = ai[v] + bi_, zf = af[v] + bf_, zg = ag[v] + bg_, zo = ao[v] + bo_;
    float cold = c[idx];
    float c2 = sigf(zf) * cold + sigf(zi) * tanhf(zg);
    float h2 = sigf(zo) * tanhf(c2);
    c[idx] = c2;
    h[idx] = h2;
  }
}

// ------------------------------------------------- inner-step row/flag update
__global__ __launch_bounds__(256) void k_post(int* __restrict__ rows,
                                              const int* __restrict__ rows_c,
                                              int* __restrict__ flags) {
  const int b = threadIdx.x;
  const int act = (flags[0] == 0);
  int r = rows[b];
  if (act) { r = rows_c[b] + 1; rows[b] = r; }
  __shared__ int nge;
  if (b == 0) nge = 0;
  __syncthreads();
  if (r >= S_LEN) atomicAdd(&nge, 1);
  __syncthreads();
  if (b == 0 && nge == B_SZ) flags[0] = 1;  // inner_done |= all(rows>=S)
}

// ----------------------------------- outer-step jump head + REINFORCE logging
__global__ __launch_bounds__(64) void k_jump(
    const float* __restrict__ h, const float* __restrict__ Wl,
    const float* __restrict__ bl, const float* __restrict__ Wb_,
    const float* __restrict__ bb_, int* __restrict__ rows,
    const int* __restrict__ rows_c, const int* __restrict__ fpflag,
    float* __restrict__ Lbuf, float* __restrict__ Bmbuf,
    float* __restrict__ step_act, const int* __restrict__ flags, int nstep) {
  const int b = blockIdx.x, tid = threadIdx.x;
  const int act = (flags[1] == 0);  // ~done
  __shared__ float slog[KP1];
  __shared__ float sbval;
  if (tid < KP1) {
    float s = bl[tid];
    const float* wrow = Wl + tid * H_DIM;
    const float* hrow = h + (size_t)b * H_DIM;
    for (int j = 0; j < H_DIM; ++j) s += wrow[j] * hrow[j];
    slog[tid] = s;
  } else if (tid == KP1) {
    float s = bb_[0];
    const float* hrow = h + (size_t)b * H_DIM;
    for (int j = 0; j < H_DIM; ++j) s += Wb_[j] * hrow[j];
    sbval = s;
  }
  __syncthreads();
  if (tid == 0) {
    float mx = slog[0];
    for (int k = 1; k < KP1; ++k) mx = fmaxf(mx, slog[k]);
    float se = 0.f;
    for (int k = 0; k < KP1; ++k) se += __expf(slog[k] - mx);
    float lse = mx + __logf(se);
    // deterministic Gumbel-max categorical sample
    int jbest = 0;
    float best = -1e30f;
    for (int k = 0; k < KP1; ++k) {
      unsigned u = hashu((unsigned)(nstep * 50021 + b * 53 + k) ^ 0xA511E9B3u);
      float uf = ((float)u + 0.5f) * (1.f / 4294967296.f);
      float v = slog[k] - __logf(-__logf(uf));
      if (v > best) { best = v; jbest = k; }
    }
    const float lp = slog[jbest] - lse;
    const int valid = act && !fpflag[b];
    Lbuf[b * NSTEPS + nstep]  = valid ? lp : 0.f;
    Bmbuf[b * NSTEPS + nstep] = valid ? sbval : 0.f;
    if (b == 0) step_act[nstep] = act ? 1.f : 0.f;
    int rn = (jbest == 0) ? S_LEN /* last_rows+1 */ : (rows_c[b] + jbest);
    if (act) rows[b] = rn;
  }
}

__global__ __launch_bounds__(256) void k_jump_post(const int* __restrict__ rows,
                                                   int* __restrict__ flags) {
  const int b = threadIdx.x;
  __shared__ int nge;
  if (b == 0) nge = 0;
  __syncthreads();
  if (rows[b] >= S_LEN) atomicAdd(&nge, 1);
  __syncthreads();
  if (b == 0) {
    if (nge == B_SZ) flags[1] = 1;  // done |= all(rows>=S)
    flags[0] = flags[1];            // inner_done = done for next outer step
  }
}

// ------------------------------------------------------ output head per batch
__global__ __launch_bounds__(64) void k_final_a(
    const float* __restrict__ h, const float* __restrict__ hiddens,
    const int* __restrict__ captured, const float* __restrict__ Wo,
    const float* __restrict__ bo, const int* __restrict__ tcat,
    float* __restrict__ nll_part, float* __restrict__ reward) {
  const int b = blockIdx.x, tid = threadIdx.x;
  __shared__ float y[2];
  if (tid < 2) {
    const float* hf = captured[b] ? (hiddens + (size_t)b * H_DIM)
                                  : (h + (size_t)b * H_DIM);
    float s = bo[tid];
    const float* w = Wo + tid * H_DIM;
    for (int j = 0; j < H_DIM; ++j) s += w[j] * hf[j];
    y[tid] = s;
  }
  __syncthreads();
  if (tid == 0) {
    float mx = fmaxf(y[0], y[1]);
    float lse = mx + __logf(__expf(y[0] - mx) + __expf(y[1] - mx));
    int t = tcat[b];
    nll_part[b] = -(y[t] - lse);
    int amax = (y[1] > y[0]) ? 1 : 0;
    reward[b] = (amax == t) ? 1.f : -1.f;
  }
}

// ---------------------------------------------------------- final reductions
__global__ __launch_bounds__(256) void k_final_b(
    const float* __restrict__ nll_part, const float* __restrict__ reward,
    const float* __restrict__ Lbuf, const float* __restrict__ Bmbuf,
    const float* __restrict__ step_act, float* __restrict__ out) {
  const int tid = threadIdx.x;
  __shared__ float red[256];
  __shared__ float nll_s, rsum_s;

  float rw = reward[tid];
  float racc = 0.f, macc = 0.f;
#pragma unroll
  for (int n = 0; n < NSTEPS; ++n) {
    float Rm = rw * step_act[n];
    float Lv = Lbuf[tid * NSTEPS + n];
    float Bv = Bmbuf[tid * NSTEPS + n];
    racc -= (Rm - Bv) * Lv;
    macc += (Bv - Rm) * (Bv - Rm);
  }
  red[tid] = nll_part[tid];
  __syncthreads();
  for (int s = 128; s > 0; s >>= 1) { if (tid < s) red[tid] += red[tid + s]; __syncthreads(); }
  if (tid == 0) nll_s = red[0] / (float)B_SZ;
  __syncthreads();
  red[tid] = racc;
  __syncthreads();
  for (int s = 128; s > 0; s >>= 1) { if (tid < s) red[tid] += red[tid + s]; __syncthreads(); }
  if (tid == 0) rsum_s = red[0];
  __syncthreads();
  red[tid] = macc;
  __syncthreads();
  for (int s = 128; s > 0; s >>= 1) { if (tid < s) red[tid] += red[tid + s]; __syncthreads(); }
  if (tid == 0) {
    float Tv = 0.f;
    for (int n = 0; n < NSTEPS; ++n) Tv += step_act[n];
    out[0] = nll_s + rsum_s / ((float)B_SZ * Tv) + red[0];
  }
}

// ============================================================================
extern "C" void kernel_launch(void* const* d_in, const int* in_sizes, int n_in,
                              void* d_out, int out_size, void* d_ws, size_t ws_size,
                              hipStream_t stream) {
  (void)in_sizes; (void)n_in; (void)out_size; (void)ws_size;
  const int*   xs      = (const int*)  d_in[0];
  /* lengths d_in[1] unused by reference */
  const int*   tcat    = (const int*)  d_in[2];
  const float* embed_w = (const float*)d_in[3];
  const float* Wih     = (const float*)d_in[4];
  const float* Whh     = (const float*)d_in[5];
  const float* bih     = (const float*)d_in[6];
  const float* bhh     = (const float*)d_in[7];
  const float* Wl      = (const float*)d_in[8];
  const float* bl      = (const float*)d_in[9];
  const float* Wb_     = (const float*)d_in[10];
  const float* bb_     = (const float*)d_in[11];
  const float* Wo      = (const float*)d_in[12];
  const float* bo      = (const float*)d_in[13];
  float* out = (float*)d_out;

  char* ws = (char*)d_ws;
  size_t o = 0;
  auto take = [&](size_t bytes) {
    size_t r = o;
    o += (bytes + 255) & ~(size_t)255;
    return r;
  };
  bf16*  Wp      = (bf16*) (ws + take((size_t)N4H * K_DIM * 2));  // 3 MB
  float* bsum    = (float*)(ws + take(N4H * 4));
  bf16*  embh    = (bf16*) (ws + take((size_t)B_SZ * K_DIM * 2));
  size_t stateBegin = o;
  float* h       = (float*)(ws + take((size_t)B_SZ * H_DIM * 4));
  float* c       = (float*)(ws + take((size_t)B_SZ * H_DIM * 4));
  float* hiddens = (float*)(ws + take((size_t)B_SZ * H_DIM * 4));
  int*   rows    = (int*)  (ws + take(B_SZ * 4));
  int*   rows_c  = (int*)  (ws + take(B_SZ * 4));
  int*   fpflag  = (int*)  (ws + take(B_SZ * 4));
  int*   captured= (int*)  (ws + take(B_SZ * 4));
  int*   flags   = (int*)  (ws + take(2 * 4));  // [0]=inner_done [1]=done
  float* Lbuf    = (float*)(ws + take(B_SZ * NSTEPS * 4));
  float* Bmbuf   = (float*)(ws + take(B_SZ * NSTEPS * 4));
  float* step_act= (float*)(ws + take(NSTEPS * 4));
  float* nll_part= (float*)(ws + take(B_SZ * 4));
  float* reward  = (float*)(ws + take(B_SZ * 4));
  long zn = (long)((o - stateBegin) / 4);

  k_zero<<<(int)((zn + 255) / 256), 256, 0, stream>>>((float*)(ws + stateBegin), zn);
  k_init_w<<<(N4H * K_DIM + 255) / 256, 256, 0, stream>>>(Wih, Whh, bih, bhh, Wp, bsum);

  for (int n = 0; n < NSTEPS; ++n) {
    for (int r = 0; r < 20; ++r) {
      k_prep<<<B_SZ, 256, 0, stream>>>(xs, embed_w, h, hiddens, rows, rows_c,
                                       fpflag, captured, embh, flags, 0);
      k_lstm_gemm<<<64, 256, 0, stream>>>(Wp, bsum, embh, h, c, flags, 0);
      k_post<<<1, 256, 0, stream>>>(rows, rows_c, flags);
    }
    // extra LSTM step gated by global `done`, then jump head
    k_prep<<<B_SZ, 256, 0, stream>>>(xs, embed_w, h, hiddens, rows, rows_c,
                                     fpflag, captured, embh, flags, 1);
    k_lstm_gemm<<<64, 256, 0, stream>>>(Wp, bsum, embh, h, c, flags, 1);
    k_jump<<<B_SZ, 64, 0, stream>>>(h, Wl, bl, Wb_, bb_, rows, rows_c, fpflag,
                                    Lbuf, Bmbuf, step_act, flags, n);
    k_jump_post<<<1, 256, 0, stream>>>(rows, flags);
  }

  k_final_a<<<B_SZ, 64, 0, stream>>>(h, hiddens, captured, Wo, bo, tcat,
                                     nll_part, reward);
  k_final_b<<<1, 256, 0, stream>>>(nll_part, reward, Lbuf, Bmbuf, step_act, out);
}
